// ArcFaceLoss_56453050139084
// MI455X (gfx1250) — compile-verified, matrix operand order checked by reading, not run on hardware
//
#include <hip/hip_runtime.h>
#include <stdint.h>

typedef __attribute__((ext_vector_type(16))) _Float16 v16h;
typedef __attribute__((ext_vector_type(8)))  float    v8f;

#define ARC_SCALE  64.0f
#define ARC_MARGIN 0.5f
#define LOG2E      1.4426950408889634f
#define SLOG2E     92.33248261689366f   // 64 * log2(e)

__device__ __forceinline__ unsigned int packh(float lo, float hi) {
    union { _Float16 h[2]; unsigned int u; } p;
    p.h[0] = (_Float16)lo;
    p.h[1] = (_Float16)hi;
    return p.u;
}

// ---------------------------------------------------------------------------
// Kernel 1: L2-normalize features -> f16 row-major [B,512]; zero rowsum.
// One wave per row; lane handles 16 contiguous dims (4x b128 loads).
// ---------------------------------------------------------------------------
__global__ __launch_bounds__(256) void knorm_kernel(const float* __restrict__ feat,
                                                    uint16_t* __restrict__ fnorm,
                                                    float* __restrict__ rowsum,
                                                    int B) {
    const int wave = threadIdx.x >> 5, lane = threadIdx.x & 31;
    const int row  = blockIdx.x * 8 + wave;
    if (row >= B) return;

    union { float4 q[4]; float f[16]; } u;
    const float4* fp = (const float4*)(feat + (size_t)row * 512 + lane * 16);
    u.q[0] = fp[0]; u.q[1] = fp[1]; u.q[2] = fp[2]; u.q[3] = fp[3];

    float s = 0.0f;
#pragma unroll
    for (int j = 0; j < 16; ++j) s += u.f[j] * u.f[j];
#pragma unroll
    for (int m = 1; m < 32; m <<= 1) s += __shfl_xor(s, m, 32);
    const float inv = 1.0f / fmaxf(sqrtf(s), 1e-12f);

    unsigned int p[8];
#pragma unroll
    for (int j = 0; j < 8; ++j) p[j] = packh(u.f[2 * j] * inv, u.f[2 * j + 1] * inv);

    uint4* dst = (uint4*)(fnorm + (size_t)row * 512 + lane * 16);
    dst[0] = make_uint4(p[0], p[1], p[2], p[3]);
    dst[1] = make_uint4(p[4], p[5], p[6], p[7]);

    if (lane == 0) rowsum[row] = 0.0f;
}

// ---------------------------------------------------------------------------
// Kernel 2: fused GEMM + partial logsumexp.
// Block = 64-class tile. Stage 1: normalize weights -> f16 tile in LDS
// (1040-byte pitch, bank-conflict free b128). Stage 2: each wave computes a
// 64-row x 64-class register block: 16 accumulators, 4 A frags (global/L2),
// B frags streamed from LDS with 4x reuse. K loop kept rolled (unroll 1) so
// nothing gets hoisted/spilled. Epilogue: exp2 + lane reduce + f32 atomics.
// ---------------------------------------------------------------------------
#define WPITCH 1040   // 512*2 bytes + 16-byte pad

__global__ __launch_bounds__(256) void kmain_kernel(const uint16_t* __restrict__ fnorm,
                                                    const float* __restrict__ weight,
                                                    float* __restrict__ rowsum,
                                                    int B, int C) {
    __shared__ __align__(16) unsigned char ldsW[64 * WPITCH];

    const int tid  = threadIdx.x;
    const int wave = tid >> 5, lane = tid & 31;
    const int half = lane >> 4, l15 = lane & 15;
    const int c0   = blockIdx.x * 64;

    // ---- Stage 1: load + normalize 64 weight rows into LDS as f16 ----
    for (int i = 0; i < 8; ++i) {
        const int c  = wave * 8 + i;       // 0..63 within tile
        const int gc = c0 + c;             // global class
        union { float4 q[4]; float f[16]; } u;
        if (gc < C) {
            const float4* wp = (const float4*)(weight + (size_t)gc * 512 + lane * 16);
            u.q[0] = wp[0]; u.q[1] = wp[1]; u.q[2] = wp[2]; u.q[3] = wp[3];
        } else {
#pragma unroll
            for (int j = 0; j < 16; ++j) u.f[j] = 0.0f;
        }
        float s = 0.0f;
#pragma unroll
        for (int j = 0; j < 16; ++j) s += u.f[j] * u.f[j];
#pragma unroll
        for (int m = 1; m < 32; m <<= 1) s += __shfl_xor(s, m, 32);
        const float inv = 1.0f / fmaxf(sqrtf(s), 1e-12f);

        unsigned int p[8];
#pragma unroll
        for (int j = 0; j < 8; ++j) p[j] = packh(u.f[2 * j] * inv, u.f[2 * j + 1] * inv);

        uint4* dst = (uint4*)(ldsW + c * WPITCH + lane * 32);
        dst[0] = make_uint4(p[0], p[1], p[2], p[3]);
        dst[1] = make_uint4(p[4], p[5], p[6], p[7]);
    }
    __syncthreads();

    // ---- Stage 2: each wave sweeps 64-row chunks of the feature matrix ----
    const v8f vzero = {0.f, 0.f, 0.f, 0.f, 0.f, 0.f, 0.f, 0.f};
    const int nchunks = B >> 6;            // 64 rows per chunk
    for (int chunk = wave; chunk < nchunks; chunk += 8) {
        const int rbase = chunk * 64;
        // A fragment addressing: lane<16 -> row rbase+l, K chunks [k,k+8) &
        // [k+16,k+24); lanes 16..31 same rows, K chunks +8 / +24.
        const uint16_t* abase = fnorm + (size_t)(rbase + l15) * 512 + half * 8;

        v8f acc[16];
#pragma unroll
        for (int i = 0; i < 16; ++i) acc[i] = vzero;

#pragma unroll 1
        for (int kk = 0; kk < 512; kk += 32) {
            union { v16h v; uint4 q[2]; } a[4];
#pragma unroll
            for (int mi = 0; mi < 4; ++mi) {
                const uint16_t* ap = abase + (size_t)(mi * 16) * 512 + kk;
                a[mi].q[0] = *(const uint4*)(ap);
                a[mi].q[1] = *(const uint4*)(ap + 16);
            }
#pragma unroll
            for (int t = 0; t < 4; ++t) {
                union { v16h v; uint4 q[2]; } bm;
                const unsigned char* bp =
                    ldsW + (t * 16 + l15) * WPITCH + (kk + half * 8) * 2;
                bm.q[0] = *(const uint4*)bp;
                bm.q[1] = *(const uint4*)(bp + 32);
#pragma unroll
                for (int mi = 0; mi < 4; ++mi)
                    acc[t * 4 + mi] = __builtin_amdgcn_wmma_f32_16x16x32_f16(
                        false, a[mi].v, false, bm.v, (short)0,
                        acc[t * 4 + mi], false, false);
            }
        }

        // Epilogue: e = exp(64*cos - 64) = exp2(SLOG2E*(cos-1)), masked pad,
        // reduce over 16 class-lanes, one f32 atomic per feature row.
#pragma unroll
        for (int mi = 0; mi < 4; ++mi) {
#pragma unroll
            for (int r = 0; r < 8; ++r) {
                float s = 0.0f;
#pragma unroll
                for (int t = 0; t < 4; ++t) {
                    const int gc = c0 + t * 16 + l15;
                    const float cosv = acc[t * 4 + mi][r];
                    s += (gc < C) ? exp2f(SLOG2E * (cosv - 1.0f)) : 0.0f;
                }
                s += __shfl_xor(s, 1, 32);
                s += __shfl_xor(s, 2, 32);
                s += __shfl_xor(s, 4, 32);
                s += __shfl_xor(s, 8, 32);
                if (l15 == 0)
                    unsafeAtomicAdd(&rowsum[rbase + mi * 16 + r + half * 8], s);
            }
        }
    }
}

// ---------------------------------------------------------------------------
// Kernel 3: label fixup. Exact cos(b, label[b]) from f32 inputs; swap the
// unmargined exp term for the margined one; record target logit.
// ---------------------------------------------------------------------------
__global__ __launch_bounds__(256) void kfix_kernel(const float* __restrict__ feat,
                                                   const float* __restrict__ weight,
                                                   const int* __restrict__ labels,
                                                   float* __restrict__ rowsum,
                                                   float* __restrict__ tgt,
                                                   int B) {
    const int wave = threadIdx.x >> 5, lane = threadIdx.x & 31;
    const int row  = blockIdx.x * 8 + wave;
    if (row >= B) return;
    const int y = labels[row];

    union { float4 q[4]; float f[16]; } uf, uw;
    const float4* fp = (const float4*)(feat + (size_t)row * 512 + lane * 16);
    const float4* wp = (const float4*)(weight + (size_t)y * 512 + lane * 16);
    uf.q[0] = fp[0]; uf.q[1] = fp[1]; uf.q[2] = fp[2]; uf.q[3] = fp[3];
    uw.q[0] = wp[0]; uw.q[1] = wp[1]; uw.q[2] = wp[2]; uw.q[3] = wp[3];

    float dot = 0.0f, ff = 0.0f, ww = 0.0f;
#pragma unroll
    for (int j = 0; j < 16; ++j) {
        dot += uf.f[j] * uw.f[j];
        ff  += uf.f[j] * uf.f[j];
        ww  += uw.f[j] * uw.f[j];
    }
#pragma unroll
    for (int m = 1; m < 32; m <<= 1) {
        dot += __shfl_xor(dot, m, 32);
        ff  += __shfl_xor(ff, m, 32);
        ww  += __shfl_xor(ww, m, 32);
    }
    if (lane == 0) {
        const float cosv = dot / (fmaxf(sqrtf(ff), 1e-12f) * fmaxf(sqrtf(ww), 1e-12f));
        const float lt   = ARC_SCALE * (cosv - ARC_MARGIN);          // target logit
        const float enew = exp2f(LOG2E * (lt - ARC_SCALE));          // exp(lt - 64)
        const float eold = exp2f(SLOG2E * (cosv - 1.0f));            // exp(64*cos - 64)
        unsafeAtomicAdd(&rowsum[row], enew - eold);
        tgt[row] = lt;
    }
}

// ---------------------------------------------------------------------------
// Kernel 4: loss = mean_b(64 + log(rowsum[b]) - tgt[b])
// ---------------------------------------------------------------------------
__global__ __launch_bounds__(256) void kloss_kernel(const float* __restrict__ rowsum,
                                                    const float* __restrict__ tgt,
                                                    float* __restrict__ out, int B) {
    __shared__ float red[8];
    float s = 0.0f;
    for (int b = threadIdx.x; b < B; b += 256)
        s += ARC_SCALE + logf(rowsum[b]) - tgt[b];
#pragma unroll
    for (int m = 1; m < 32; m <<= 1) s += __shfl_xor(s, m, 32);
    if ((threadIdx.x & 31) == 0) red[threadIdx.x >> 5] = s;
    __syncthreads();
    if (threadIdx.x == 0) {
        float t = 0.0f;
        for (int i = 0; i < 8; ++i) t += red[i];
        out[0] = t / (float)B;
    }
}

extern "C" void kernel_launch(void* const* d_in, const int* in_sizes, int n_in,
                              void* d_out, int out_size, void* d_ws, size_t ws_size,
                              hipStream_t stream) {
    const float* feat   = (const float*)d_in[0];
    const int*   labels = (const int*)d_in[1];
    const float* weight = (const float*)d_in[2];
    const int B = in_sizes[1];            // 1024
    const int C = in_sizes[2] / 512;      // 100000

    uint16_t* fnorm  = (uint16_t*)d_ws;                          // B*512 f16
    float*    rowsum = (float*)((char*)d_ws + (size_t)B * 512 * 2);
    float*    tgt    = rowsum + B;
    float*    out    = (float*)d_out;

    knorm_kernel<<<B / 8, 256, 0, stream>>>(feat, fnorm, rowsum, B);
    kmain_kernel<<<(C + 63) / 64, 256, 0, stream>>>(fnorm, weight, rowsum, B, C);
    kfix_kernel<<<B / 8, 256, 0, stream>>>(feat, weight, labels, rowsum, tgt, B);
    kloss_kernel<<<1, 256, 0, stream>>>(rowsum, tgt, out, B);
}
